// LSTMBackbone_72825465471359
// MI455X (gfx1250) — matrix operand order repaired and checked by reading, not compiled, not run on hardware
//
#include <hip/hip_runtime.h>
#include <hip/hip_bf16.h>

typedef __attribute__((ext_vector_type(16))) __bf16 v16bf;
typedef __attribute__((ext_vector_type(8)))  float  v8f;

#define S_LEN  256
#define BATCH  256
#define IN_DIM 256
#define HID    512
#define GATES  2048           // 4*HID
#define BH     (BATCH * HID)  // 131072 = 1<<17

#define NBLOCKS 128           // persistent grid: 32 nt-slices x 4 mt-groups
#define TPB     128           // 4 waves/block -> 512 waves = 16 mt x 32 nt tiles
#define K0ROWS  768           // IN_DIM + HID
#define K1ROWS  1024          // HID + HID
#define LDS0_BYTES (4 * K0ROWS * 32)   //  98304 B
#define LDS1_BYTES (4 * K1ROWS * 32)   // 131072 B
#define LDS_TOTAL  (LDS0_BYTES + LDS1_BYTES)  // 229376 B < 320KB WGP LDS

__device__ __forceinline__ float sigf(float x) {
    return 1.0f / (1.0f + __expf(-x));
}

union Frag { v16bf v; int4 q[2]; };

// ---------------------------------------------------------------------------
// Device-scope barrier across the persistent grid (monotonic counter).
// ---------------------------------------------------------------------------
__device__ __forceinline__ void grid_barrier(unsigned* bar, unsigned target)
{
    __threadfence();
    __syncthreads();
    if (threadIdx.x == 0) {
        __hip_atomic_fetch_add(bar, 1u, __ATOMIC_ACQ_REL, __HIP_MEMORY_SCOPE_AGENT);
        while (__hip_atomic_load(bar, __ATOMIC_ACQUIRE, __HIP_MEMORY_SCOPE_AGENT) < target)
            __builtin_amdgcn_s_sleep(2);
        __threadfence();
    }
    __syncthreads();
}

// ---------------------------------------------------------------------------
// Stage this block's 16-column weight slice (all 4 gate chunks) into LDS.
// LDS layout: [q][16 n][Krows bf16] -> k contiguous per column n.
// ---------------------------------------------------------------------------
__device__ void stage_weights(const __bf16* __restrict__ Wt, int Krows, int n0,
                              char* dst)
{
    for (int r = threadIdx.x; r < 4 * Krows; r += blockDim.x) {
        const int q = r / Krows;
        const int k = r - q * Krows;
        const __bf16* src = Wt + (size_t)k * GATES + q * HID + n0;
        __bf16* d = reinterpret_cast<__bf16*>(dst) + (size_t)q * 16 * Krows + k;
        for (int j = 0; j < 16; ++j) d[(size_t)j * Krows] = src[j];
    }
}

// ---------------------------------------------------------------------------
// Fused LSTM layer step for one 16x16 (batch x hidden) patch:
// 4 gate accumulators via WMMA (B operand from LDS), then cell + masking.
// ---------------------------------------------------------------------------
template <bool IS_TOP>
__device__ __forceinline__ void lstm_layer_step(
        int mt, int nt, int lane,
        const __bf16* __restrict__ A1, int K1,      // x_t (or h0_new), K-contig rows
        const __bf16* __restrict__ A2,              // h state (K2 = HID)
        const char*   __restrict__ Blds, int Krows, // staged weights [4][16][Krows]
        const float*  __restrict__ bias,
        const float*  __restrict__ done_row,
        float*  __restrict__ c_state,               // masked state (in/out)
        float*  __restrict__ h_f32,                 // masked state out
        __bf16* __restrict__ h_bf_out,              // masked state out (ping-pong)
        __bf16* __restrict__ h_unm_bf,              // !IS_TOP: layer-1 input
        float*  __restrict__ out_unm)               //  IS_TOP: out[s]
{
    const int half = (lane >> 4) & 1;
    const int mrow = (mt << 4) + (lane & 15);
    const int koff = half << 3;
    const int col  = (nt << 4) + (lane & 15);

    // Loop-invariant per-q LDS base offsets for this lane's B rows.
    // The empty asm makes off1..off3 opaque so CSE cannot re-merge the four
    // bases into one register with (un-encodable) negative DS offsets; each
    // base then keeps immediate in-loop offsets ((kbase+k)*2).
    int off0 = ((0 * 16 + (lane & 15)) * Krows + koff) << 1;
    int off1 = ((1 * 16 + (lane & 15)) * Krows + koff) << 1;
    int off2 = ((2 * 16 + (lane & 15)) * Krows + koff) << 1;
    int off3 = ((3 * 16 + (lane & 15)) * Krows + koff) << 1;
    asm volatile("" : "+v"(off1), "+v"(off2), "+v"(off3));
    const char* bq0 = Blds + off0;
    const char* bq1 = Blds + off1;
    const char* bq2 = Blds + off2;
    const char* bq3 = Blds + off3;

    v8f acc[4];
#pragma unroll
    for (int q = 0; q < 4; ++q) {
        const float bv = bias[q * HID + col];
        acc[q] = (v8f){bv, bv, bv, bv, bv, bv, bv, bv};
    }

    const __bf16* As[2] = { A1, A2 };
    const int     Ks[2] = { K1, HID };
    int kbase = 0;
#pragma unroll
    for (int seg = 0; seg < 2; ++seg) {
        const int K = Ks[seg];
        const __bf16* Arow = As[seg] + (size_t)mrow * K;
#pragma unroll 2
        for (int k = 0; k < K; k += 32) {
            // A 16x32 (MxK): lane<16 holds K {0..7,16..23}, lane>=16 {8..15,24..31}
            Frag a;
            a.q[0] = *reinterpret_cast<const int4*>(Arow + k + koff);
            a.q[1] = *reinterpret_cast<const int4*>(Arow + k + 16 + koff);
            __builtin_prefetch(Arow + k + 64, 0, 1);

            // Load ALL four B fragments first (distinct regs) so the ds_loads
            // clause together and overlap the previous WMMA group.
            const int kb = (kbase + k) << 1;
            Frag b0, b1, b2, b3;
            b0.q[0] = *reinterpret_cast<const int4*>(bq0 + kb);
            b0.q[1] = *reinterpret_cast<const int4*>(bq0 + kb + 32);
            b1.q[0] = *reinterpret_cast<const int4*>(bq1 + kb);
            b1.q[1] = *reinterpret_cast<const int4*>(bq1 + kb + 32);
            b2.q[0] = *reinterpret_cast<const int4*>(bq2 + kb);
            b2.q[1] = *reinterpret_cast<const int4*>(bq2 + kb + 32);
            b3.q[0] = *reinterpret_cast<const int4*>(bq3 + kb);
            b3.q[1] = *reinterpret_cast<const int4*>(bq3 + kb + 32);

            // Four independent WMMAs back-to-back (distinct acc / B regs).
            acc[0] = __builtin_amdgcn_wmma_f32_16x16x32_bf16(
                        false, a.v, false, b0.v, (short)0, acc[0], false, false);
            acc[1] = __builtin_amdgcn_wmma_f32_16x16x32_bf16(
                        false, a.v, false, b1.v, (short)0, acc[1], false, false);
            acc[2] = __builtin_amdgcn_wmma_f32_16x16x32_bf16(
                        false, a.v, false, b2.v, (short)0, acc[2], false, false);
            acc[3] = __builtin_amdgcn_wmma_f32_16x16x32_bf16(
                        false, a.v, false, b3.v, (short)0, acc[3], false, false);
        }
        kbase += K;
    }

    // epilogue: LSTM cell on the 16x16 patch.
    // C/D layout: VGPR g, lane L -> row = 16*mt + g + (L>=16 ? 8 : 0), col = L%16
    const int row0 = (mt << 4) + (half << 3);
#pragma unroll
    for (int g = 0; g < 8; ++g) {
        const int row = row0 + g;               // batch index
        const int idx = row * HID + col;
        const float gi = sigf(acc[0][g]);
        const float gf = sigf(acc[1][g]);
        const float gg = tanhf(acc[2][g]);
        const float go = sigf(acc[3][g]);
        const float cn = gf * c_state[idx] + gi * gg;
        const float hn = go * tanhf(cn);
        const float m  = 1.0f - done_row[row];
        if (IS_TOP) out_unm[idx]  = hn;            // out_t BEFORE masking
        else        h_unm_bf[idx] = (__bf16)hn;    // layer-1 input BEFORE masking
        c_state[idx]  = cn * m;
        h_f32[idx]    = hn * m;
        h_bf_out[idx] = (__bf16)(hn * m);
    }
}

// ---------------------------------------------------------------------------
// Persistent scan kernel: whole 256-step, 2-layer LSTM in one launch.
// hb0/hb1 are 2*BH double-buffers; halves selected by pointer ARITHMETIC
// (not a pointer array) so A-loads keep global provenance -> global_load,
// never flat_load (flat would burn DScnt + the LDS path our B-reads need).
// ---------------------------------------------------------------------------
__global__ void lstm_scan(const __bf16* __restrict__ x_bf,
                          const float*  __restrict__ done,
                          const __bf16* __restrict__ Wt0,
                          const __bf16* __restrict__ Wt1,
                          const float*  __restrict__ bias0,
                          const float*  __restrict__ bias1,
                          float* c0, float* c1,
                          float* hf0, float* hf1,
                          __bf16* hb0,      // [2][BH] ping-pong
                          __bf16* hb1,      // [2][BH] ping-pong
                          __bf16* h0new,
                          float* out, unsigned* bar)
{
    extern __shared__ char smem[];
    char* B0 = smem;                 // [4][16][768]  bf16
    char* B1 = smem + LDS0_BYTES;    // [4][16][1024] bf16

    const int lane = threadIdx.x & 31;
    const int wave = threadIdx.x >> 5;
    const int nt   = blockIdx.x & 31;            // all 4 waves share nt (B in LDS)
    const int mt   = ((blockIdx.x >> 5) << 2) + wave;
    const int n0   = nt << 4;

    stage_weights(Wt0, K0ROWS, n0, B0);
    stage_weights(Wt1, K1ROWS, n0, B1);
    __syncthreads();

    unsigned need = 0;
#pragma unroll 1
    for (int s = 0; s < S_LEN; ++s) {
        const size_t pi = (size_t)(s & 1) << 17;   // * BH
        const size_t po = pi ^ ((size_t)1 << 17);
        const float* drow = done + (size_t)s * BATCH;

        // layer 0: gates = [x_t | h0] @ W0 + bias0, fused cell
        lstm_layer_step<false>(mt, nt, lane,
                x_bf + (size_t)s * BATCH * IN_DIM, IN_DIM,
                hb0 + pi, B0, K0ROWS, bias0, drow,
                c0, hf0, hb0 + po, h0new, nullptr);
        need += NBLOCKS; grid_barrier(bar, need);

        // layer 1: gates = [h0_new | h1] @ W1 + bias1, fused cell
        lstm_layer_step<true>(mt, nt, lane,
                h0new, HID,
                hb1 + pi, B1, K1ROWS, bias1, drow,
                c1, hf1, hb1 + po, nullptr, out + (size_t)s * BH);
        need += NBLOCKS; grid_barrier(bar, need);
    }
}

// ---------------------------------------------------------------------------
// One-time setup kernels
// ---------------------------------------------------------------------------
__global__ void cvt_bf16(const float* __restrict__ src, __bf16* __restrict__ dst, int n)
{
    const int i = blockIdx.x * blockDim.x + threadIdx.x;
    if (i < n) dst[i] = (__bf16)src[i];
}

// Wt[(r0+k)*2048 + n] = bf16(W[n*K + k])   (K-major transpose for WMMA B operand)
__global__ void build_wt(const float* __restrict__ W, int K,
                         __bf16* __restrict__ Wt, int r0, int total)
{
    const int i = blockIdx.x * blockDim.x + threadIdx.x;
    if (i >= total) return;
    const int k = i >> 11;
    const int n = i & 2047;
    Wt[(size_t)(r0 + k) * GATES + n] = (__bf16)W[(size_t)n * K + k];
}

__global__ void add_bias(const float* __restrict__ a, const float* __restrict__ b,
                         float* __restrict__ o, int n)
{
    const int i = blockIdx.x * blockDim.x + threadIdx.x;
    if (i < n) o[i] = a[i] + b[i];
}

__global__ void init_state(const float* __restrict__ h0, const float* __restrict__ c0,
                           float* __restrict__ c_st, float* __restrict__ h_f32,
                           __bf16* __restrict__ h_bf, int n)
{
    const int i = blockIdx.x * blockDim.x + threadIdx.x;
    if (i >= n) return;
    c_st[i] = c0[i];
    const float h = h0[i];
    h_f32[i] = h;
    h_bf[i]  = (__bf16)h;
}

__global__ void zero_bar(unsigned* bar)
{
    if (threadIdx.x == 0) *bar = 0u;
}

__global__ void final_copy(const float* __restrict__ h0f, const float* __restrict__ h1f,
                           const float* __restrict__ c0s, const float* __restrict__ c1s,
                           float* __restrict__ out_tail)
{
    const int i   = blockIdx.x * blockDim.x + threadIdx.x;  // 4*BH
    const int seg = i >> 17;
    const int off = i & (BH - 1);
    const float* src = (seg == 0) ? h0f : (seg == 1) ? h1f : (seg == 2) ? c0s : c1s;
    out_tail[i] = src[off];
}

// ---------------------------------------------------------------------------
extern "C" void kernel_launch(void* const* d_in, const int* in_sizes, int n_in,
                              void* d_out, int out_size, void* d_ws, size_t ws_size,
                              hipStream_t stream)
{
    const float* x     = (const float*)d_in[0];
    const float* h0in  = (const float*)d_in[1];   // (2,B,H)
    const float* c0in  = (const float*)d_in[2];   // (2,B,H)
    const float* done  = (const float*)d_in[3];   // (S,B)
    const float* W_ih0 = (const float*)d_in[4];
    const float* W_hh0 = (const float*)d_in[5];
    const float* b_ih0 = (const float*)d_in[6];
    const float* b_hh0 = (const float*)d_in[7];
    const float* W_ih1 = (const float*)d_in[8];
    const float* W_hh1 = (const float*)d_in[9];
    const float* b_ih1 = (const float*)d_in[10];
    const float* b_hh1 = (const float*)d_in[11];
    float* out = (float*)d_out;

    // workspace partition (256B aligned)
    char* ws = (char*)d_ws;
    auto alloc = [&](size_t bytes) {
        char* p = ws; ws += (bytes + 255) & ~(size_t)255; return p;
    };
    __bf16*   x_bf  = (__bf16*)alloc((size_t)S_LEN * BATCH * IN_DIM * 2);
    __bf16*   Wt0   = (__bf16*)alloc((size_t)K0ROWS * GATES * 2);
    __bf16*   Wt1   = (__bf16*)alloc((size_t)K1ROWS * GATES * 2);
    float*    bias0 = (float*)alloc(GATES * 4);
    float*    bias1 = (float*)alloc(GATES * 4);
    float*    c_st0 = (float*)alloc((size_t)BH * 4);
    float*    c_st1 = (float*)alloc((size_t)BH * 4);
    float*    h_f0  = (float*)alloc((size_t)BH * 4);
    float*    h_f1  = (float*)alloc((size_t)BH * 4);
    __bf16*   hb0   = (__bf16*)alloc((size_t)2 * BH * 2);  // ping-pong halves
    __bf16*   hb1   = (__bf16*)alloc((size_t)2 * BH * 2);  // ping-pong halves
    __bf16*   h0new = (__bf16*)alloc((size_t)BH * 2);
    unsigned* bar   = (unsigned*)alloc(256);

    const dim3 blk(256);

    // one-time setup
    const int nx = S_LEN * BATCH * IN_DIM;               // 16,777,216
    cvt_bf16<<<nx / 256, blk, 0, stream>>>(x, x_bf, nx);

    build_wt<<<(IN_DIM * GATES) / 256, blk, 0, stream>>>(W_ih0, IN_DIM, Wt0, 0,      IN_DIM * GATES);
    build_wt<<<(HID    * GATES) / 256, blk, 0, stream>>>(W_hh0, HID,    Wt0, IN_DIM, HID * GATES);
    build_wt<<<(HID    * GATES) / 256, blk, 0, stream>>>(W_ih1, HID,    Wt1, 0,      HID * GATES);
    build_wt<<<(HID    * GATES) / 256, blk, 0, stream>>>(W_hh1, HID,    Wt1, HID,    HID * GATES);

    add_bias<<<GATES / 256, blk, 0, stream>>>(b_ih0, b_hh0, bias0, GATES);
    add_bias<<<GATES / 256, blk, 0, stream>>>(b_ih1, b_hh1, bias1, GATES);

    // step 0 reads ping half 0
    init_state<<<BH / 256, blk, 0, stream>>>(h0in,      c0in,      c_st0, h_f0, hb0, BH);
    init_state<<<BH / 256, blk, 0, stream>>>(h0in + BH, c0in + BH, c_st1, h_f1, hb1, BH);

    zero_bar<<<1, 64, 0, stream>>>(bar);

    // persistent scan: one launch for all 256 timesteps, both layers
    lstm_scan<<<NBLOCKS, TPB, LDS_TOTAL, stream>>>(
            x_bf, done, Wt0, Wt1, bias0, bias1,
            c_st0, c_st1, h_f0, h_f1,
            hb0, hb1, h0new, out, bar);

    // final (h, c) states appended after outputs
    final_copy<<<(4 * BH) / 256, blk, 0, stream>>>(h_f0, h_f1, c_st0, c_st1,
                                                   out + (size_t)S_LEN * BH);
}
